// DenseRecurrentConsciousnessNet_78108275245217
// MI455X (gfx1250) — matrix operand order repaired and checked by reading, not compiled, not run on hardware
//
#include <hip/hip_runtime.h>

typedef unsigned short ushort_t;
typedef __attribute__((ext_vector_type(16))) __bf16 v16bf;
typedef __attribute__((ext_vector_type(8)))  float  v8f;

static constexpr int kB = 65536;  // batch
static constexpr int kH = 512;    // hidden
static constexpr int kS = 256;    // memory slots
static constexpr int kD = 64;     // slot size

// LDS strides (in ushort elements). All chosen so that:
//  - stride*2 is a multiple of 16 bytes (aligned ds_load_b128)
//  - (stride*2/4) % 64 gives distinct banks across the 16 lanes of a fragment
static constexpr int QS_STRIDE = 40;   // q chunk [64][32] padded
static constexpr int WT_STRIDE = 40;   // transposed W chunk [256][32] padded
static constexpr int PR_STRIDE = 264;  // read-probs [64][256] padded
static constexpr int MT_STRIDE = 264;  // transposed memory [64][256] padded

struct Frag32B { uint4 lo, hi; };

__device__ __forceinline__ ushort_t f2bf(float f) {
  unsigned u = __float_as_uint(f);
  u += 0x7FFFu + ((u >> 16) & 1u);   // round-to-nearest-even
  return (ushort_t)(u >> 16);
}

// A-matrix fragment (16x32 bf16), row-major source in LDS.
// ISA layout: lanes 0-15: M=lane, K = 0..7 (v0..3) and 16..23 (v4..7);
//             lanes 16-31: M=lane-16, K = 8..15 and 24..31.
__device__ __forceinline__ v16bf load_a(const ushort_t* base, int stride, int k0, int lane) {
  const int m  = lane & 15;
  const int kb = (lane >> 4) << 3;            // 0 or 8
  const ushort_t* p = base + m * stride + k0 + kb;
  Frag32B f;
  f.lo = *(const uint4*)(p);                   // K = k0+kb .. +7
  f.hi = *(const uint4*)(p + 16);              // K = k0+16+kb .. +7
  return __builtin_bit_cast(v16bf, f);
}

// B-matrix fragment (32x16 bf16) from a K-contiguous (transposed) LDS layout
// baseT[n][k]. ISA layout: lanes 0-15 hold K=0..7,16..23; lanes 16-31 K=8..15,24..31.
__device__ __forceinline__ v16bf load_bT(const ushort_t* baseT, int stride, int n0, int k0, int lane) {
  const int n  = lane & 15;
  const int kb = (lane >> 4) << 3;            // 0 or 8
  const ushort_t* p = baseT + (n0 + n) * stride + k0 + kb;
  Frag32B f;
  f.lo = *(const uint4*)(p);
  f.hi = *(const uint4*)(p + 16);
  return __builtin_bit_cast(v16bf, f);
}

__global__ void init_ws(float* __restrict__ w_sum, float* __restrict__ c_sum) {
  const int t = threadIdx.x;              // 256 threads
  w_sum[t] = 0.f;
  c_sum[t] = 0.f;
  c_sum[t + 256] = 0.f;
}

// Column sums of content [B,H] -> c_sum[H]. 1024 blocks x 256 threads, 64 rows each.
__global__ __launch_bounds__(256) void content_colsum(const float* __restrict__ content,
                                                      float* __restrict__ c_sum) {
  const int t = threadIdx.x;
  const int row0 = blockIdx.x * 64;
  float a0 = 0.f, a1 = 0.f;
  for (int r = 0; r < 64; ++r) {
    const float* p = content + (size_t)(row0 + r) * kH;
    a0 += p[t];
    a1 += p[t + 256];
  }
  atomicAdd(&c_sum[t], a0);
  atomicAdd(&c_sum[t + 256], a1);
}

// Fused: logits (read+write) via WMMA, in-wave softmax, read_content via WMMA,
// write-prob column sums. 64 rows per block, 8 waves.
__global__ __launch_bounds__(256) void fused_main(const float* __restrict__ query,
                                                  const float* __restrict__ Wr,
                                                  const float* __restrict__ br,
                                                  const float* __restrict__ Ww,
                                                  const float* __restrict__ bw,
                                                  const float* __restrict__ memv,
                                                  float* __restrict__ out_read,
                                                  float* __restrict__ w_sum_g) {
  // Region X (phase 1): q chunk + transposed W chunks; (phase 3): transposed memory.
  __shared__ __align__(16) ushort_t smemX[64 * QS_STRIDE + 2 * kS * WT_STRIDE]; // 46080 B
  // Region Y: bf16 read-probs, written in softmax, consumed in phase 3.
  __shared__ __align__(16) ushort_t smemY[64 * PR_STRIDE];                      // 33792 B
  __shared__ float ws_col[kS];                                                  // 1 KB

  ushort_t* qsc  = smemX;                       // [64][QS_STRIDE]
  ushort_t* wrT  = smemX + 64 * QS_STRIDE;      // [256][WT_STRIDE]
  ushort_t* wwT  = wrT + kS * WT_STRIDE;        // [256][WT_STRIDE]
  ushort_t* memT = smemX;                       // [64][MT_STRIDE] (reuses phase-1 space)
  ushort_t* pr   = smemY;                       // [64][PR_STRIDE]

  const int tid  = threadIdx.x;
  const int wave = tid >> 5;
  const int lane = tid & 31;
  const int row0 = blockIdx.x * 64;

  if (tid < kS) ws_col[tid] = 0.f;

  const int mat = wave >> 2;                    // 0 = read, 1 = write
  const int rt  = wave & 3;                     // row tile (16 rows)
  const ushort_t* myW   = mat ? wwT : wrT;
  const ushort_t* aBase = qsc + (rt * 16) * QS_STRIDE;

  const v8f vzero = {0.f, 0.f, 0.f, 0.f, 0.f, 0.f, 0.f, 0.f};
  v8f acc[16];
  #pragma unroll
  for (int j = 0; j < 16; ++j) acc[j] = vzero;

  // ---- Phase 1: logits = q @ W  (K = 512 in 16 chunks of 32) ----
  for (int kc = 0; kc < 16; ++kc) {
    const int k0 = kc * 32;
    // stage q chunk (bf16)
    for (int i = tid; i < 64 * 32; i += 256) {
      const int r = i >> 5, k = i & 31;
      qsc[r * QS_STRIDE + k] = f2bf(query[(row0 + r) * kH + k0 + k]);
    }
    // stage both W chunks, transposed to K-contiguous bf16
    for (int i = tid; i < 32 * kS; i += 256) {
      const int k = i >> 8, n = i & (kS - 1);
      wrT[n * WT_STRIDE + k] = f2bf(Wr[(k0 + k) * kS + n]);
      wwT[n * WT_STRIDE + k] = f2bf(Ww[(k0 + k) * kS + n]);
    }
    __syncthreads();
    const v16bf af = load_a(aBase, QS_STRIDE, 0, lane);
    // Batch 4 B-fragments per group: all four are live at the first WMMA of the
    // group, forcing 4 distinct VGPR octets so the backend can retire them with
    // partial s_wait_dscnt values and overlap WMMAs with outstanding ds_loads.
    #pragma unroll
    for (int g = 0; g < 4; ++g) {
      const v16bf b0 = load_bT(myW, WT_STRIDE, (g * 4 + 0) * 16, 0, lane);
      const v16bf b1 = load_bT(myW, WT_STRIDE, (g * 4 + 1) * 16, 0, lane);
      const v16bf b2 = load_bT(myW, WT_STRIDE, (g * 4 + 2) * 16, 0, lane);
      const v16bf b3 = load_bT(myW, WT_STRIDE, (g * 4 + 3) * 16, 0, lane);
      acc[g * 4 + 0] = __builtin_amdgcn_wmma_f32_16x16x32_bf16(false, af, false, b0,
                                                               (short)0, acc[g * 4 + 0], false, false);
      acc[g * 4 + 1] = __builtin_amdgcn_wmma_f32_16x16x32_bf16(false, af, false, b1,
                                                               (short)0, acc[g * 4 + 1], false, false);
      acc[g * 4 + 2] = __builtin_amdgcn_wmma_f32_16x16x32_bf16(false, af, false, b2,
                                                               (short)0, acc[g * 4 + 2], false, false);
      acc[g * 4 + 3] = __builtin_amdgcn_wmma_f32_16x16x32_bf16(false, af, false, b3,
                                                               (short)0, acc[g * 4 + 3], false, false);
    }
    __syncthreads();
  }

  // ---- Phase 2: bias + in-wave softmax over S=256 ----
  const float* bias = mat ? bw : br;
  const int n  = lane & 15;
  const int mb = (lane >> 4) << 3;              // row base inside the 16-row tile
  #pragma unroll
  for (int j = 0; j < 16; ++j) {
    const float bv = bias[j * 16 + n];
    #pragma unroll
    for (int v = 0; v < 8; ++v) acc[j][v] += bv;
  }

  #pragma unroll
  for (int v = 0; v < 8; ++v) {
    float mx = acc[0][v];
    #pragma unroll
    for (int j = 1; j < 16; ++j) mx = fmaxf(mx, acc[j][v]);
    #pragma unroll
    for (int m = 8; m >= 1; m >>= 1) mx = fmaxf(mx, __shfl_xor(mx, m, 32));
    float e[16], s = 0.f;
    #pragma unroll
    for (int j = 0; j < 16; ++j) { e[j] = __expf(acc[j][v] - mx); s += e[j]; }
    #pragma unroll
    for (int m = 8; m >= 1; m >>= 1) s += __shfl_xor(s, m, 32);
    const float inv = 1.0f / s;
    if (mat == 0) {
      const int gr = rt * 16 + mb + v;
      #pragma unroll
      for (int j = 0; j < 16; ++j) pr[gr * PR_STRIDE + j * 16 + n] = f2bf(e[j] * inv);
    } else {
      #pragma unroll
      for (int j = 0; j < 16; ++j) acc[j][v] = e[j] * inv;
    }
  }

  // write-prob column sums (16 rows per wave) -> LDS
  if (mat == 1) {
    #pragma unroll
    for (int j = 0; j < 16; ++j) {
      float s2 = 0.f;
      #pragma unroll
      for (int v = 0; v < 8; ++v) s2 += acc[j][v];
      s2 += __shfl_xor(s2, 16, 32);             // combine both 8-row halves
      if (lane < 16) atomicAdd(&ws_col[j * 16 + n], s2);
    }
  }

  // stage memory transposed as bf16 (overwrites phase-1 staging space)
  for (int i = tid; i < kS * kD; i += 256) {
    const int s = i >> 6, d = i & 63;
    memT[d * MT_STRIDE + s] = f2bf(memv[i]);
  }
  __syncthreads();

  // flush per-block write-prob sums (one atomic per slot per block)
  if (tid < kS) atomicAdd(&w_sum_g[tid], ws_col[tid]);

  // ---- Phase 3: read_content = p_read @ memory  (M=64, N=64, K=256) ----
  #pragma unroll
  for (int q2 = 0; q2 < 2; ++q2) {
    const int job = wave * 2 + q2;
    const int rt3 = job >> 2, ct3 = job & 3;
    const ushort_t* aB = pr + (rt3 * 16) * PR_STRIDE;
    v8f acc2 = vzero;
    // pipeline A/B fragment loads one K-step ahead
    v16bf af  = load_a(aB, PR_STRIDE, 0, lane);
    v16bf bfr = load_bT(memT, MT_STRIDE, ct3 * 16, 0, lane);
    #pragma unroll
    for (int kc = 0; kc < 8; ++kc) {
      v16bf afn = af, bfn = bfr;
      if (kc + 1 < 8) {
        afn = load_a(aB, PR_STRIDE, (kc + 1) * 32, lane);
        bfn = load_bT(memT, MT_STRIDE, ct3 * 16, (kc + 1) * 32, lane);
      }
      acc2 = __builtin_amdgcn_wmma_f32_16x16x32_bf16(false, af, false, bfr,
                                                     (short)0, acc2, false, false);
      af = afn; bfr = bfn;
    }
    #pragma unroll
    for (int v = 0; v < 8; ++v) {
      const int row = row0 + rt3 * 16 + mb + v;
      out_read[(size_t)row * kD + ct3 * 16 + n] = acc2[v];
    }
  }
}

// Tiny epilogue: c_mean = (c_sum/B)@W_content + b_content; EMA memory update; age.
__global__ __launch_bounds__(256) void finalize(const float* __restrict__ memv,
                                                const float* __restrict__ age,
                                                const float* __restrict__ Wc,
                                                const float* __restrict__ bc,
                                                const float* __restrict__ w_sum,
                                                const float* __restrict__ c_sum,
                                                float* __restrict__ out) {
  __shared__ float cm[kD];
  const int t = threadIdx.x;
  if (t < kD) {
    float a = 0.f;
    for (int h = 0; h < kH; ++h) a += c_sum[h] * Wc[h * kD + t];
    cm[t] = a * (1.0f / (float)kB) + bc[t];
  }
  __syncthreads();
  const float wm = w_sum[t] * (1.0f / (float)kB);
  const bool active = wm > 0.01f;
  const float cons = 1.0f / (1.0f + __expf(-age[t] * 0.1f));
  const float alpha = active ? wm * cons : 0.f;
  const size_t OUT1 = (size_t)kB * kD;             // new_memory offset
  const size_t OUT2 = OUT1 + (size_t)kS * kD;      // new_age offset
  out[OUT2 + t] = age[t] + (active ? 1.0f : 0.0f);
  #pragma unroll
  for (int d = 0; d < kD; ++d)
    out[OUT1 + (size_t)t * kD + d] = (1.0f - alpha) * memv[t * kD + d] + alpha * cm[d];
}

extern "C" void kernel_launch(void* const* d_in, const int* in_sizes, int n_in,
                              void* d_out, int out_size, void* d_ws, size_t ws_size,
                              hipStream_t stream) {
  const float* query   = (const float*)d_in[0];
  const float* content = (const float*)d_in[1];
  const float* memv    = (const float*)d_in[2];
  const float* age     = (const float*)d_in[3];
  const float* Wr      = (const float*)d_in[4];
  const float* br      = (const float*)d_in[5];
  const float* Ww      = (const float*)d_in[6];
  const float* bw      = (const float*)d_in[7];
  const float* Wc      = (const float*)d_in[8];
  const float* bc      = (const float*)d_in[9];
  float* out = (float*)d_out;

  float* w_sum = (float*)d_ws;        // [256]
  float* c_sum = w_sum + kS;          // [512]

  init_ws<<<1, 256, 0, stream>>>(w_sum, c_sum);
  content_colsum<<<kB / 64, 256, 0, stream>>>(content, c_sum);
  fused_main<<<kB / 64, 256, 0, stream>>>(query, Wr, br, Ww, bw, memv, out, w_sum);
  finalize<<<1, 256, 0, stream>>>(memv, age, Wc, bc, w_sum, c_sum, out);
}
